// Autoencoder_39316130628143
// MI455X (gfx1250) — matrix operand order repaired and checked by reading, not compiled, not run on hardware
//
#include <hip/hip_runtime.h>
#include <stdint.h>

// ---------------------------------------------------------------------------
// Problem constants (match reference)
// ---------------------------------------------------------------------------
static constexpr int B  = 4096;
static constexpr int D  = 1280;    // K-dim of encode GEMM
static constexpr int L  = 16384;   // N-dim of encode GEMM
static constexpr int KS = 64;      // top-k
static constexpr int KT = D / 32;  // 40 K-tiles of 32 (bf16 WMMA K)

typedef __attribute__((ext_vector_type(16))) __bf16 v16bf;
typedef __attribute__((ext_vector_type(8)))  float  v8f;

// Async copy: 16 bytes/lane, global -> LDS, ASYNCcnt-tracked (CDNA5 §15.18 op 98)
#define ASYNC_CP16(ldsoff, gptr)                                                 \
    asm volatile("global_load_async_to_lds_b128 %0, %1, off"                     \
                 :: "v"(ldsoff), "v"((unsigned long long)(uintptr_t)(gptr))      \
                 : "memory")
#define WAIT_ASYNC0() asm volatile("s_wait_asynccnt 0x0" ::: "memory")

// ---------------------------------------------------------------------------
// prep_w: W_enc [D,L] f32 -> (hi,lo) bf16 in WMMA B-fragment swizzled tiles.
// B-matrix 32x16 (KxN) lane order used here (16-bit, contiguous k):
//   lanes 0-15 : N = lane,    K = 0..15  (VGPR v holds K=2v,2v+1)
//   lanes 16-31: N = lane-16, K = 16..31
// Tile storage: tile (nt, kt) at element offset (nt*KT + kt)*512,
// within-tile element s = lane*16 + q  ->  n = nt*16 + (lane&15),
//                                          k = kt*32 + (lane>=16?16:0) + q
// ---------------------------------------------------------------------------
__global__ __launch_bounds__(256) void prep_w(const float* __restrict__ W,
                                              __bf16* __restrict__ Wh,
                                              __bf16* __restrict__ Wl) {
    size_t idx = (size_t)blockIdx.x * 256 + threadIdx.x;   // over D*L
    size_t tile = idx >> 9;
    int s    = (int)(idx & 511);
    int lane = s >> 4, q = s & 15;
    int nt = (int)(tile / KT), kt = (int)(tile % KT);
    int n = nt * 16 + (lane & 15);
    int k = kt * 32 + ((lane >> 4) << 4) + q;
    float w = W[(size_t)k * L + n];
    __bf16 h = (__bf16)w;
    Wh[idx] = h;
    Wl[idx] = (__bf16)(w - (float)h);
}

// ---------------------------------------------------------------------------
// prep_x: (x - pb) [B,D] f32 -> (hi,lo) bf16 in WMMA A-fragment swizzled tiles.
// A-matrix 16x32 (MxK) lane order used here (16-bit, contiguous k):
//   lanes 0-15 : M = lane,    K = 0..15
//   lanes 16-31: M = lane-16, K = 16..31
// ---------------------------------------------------------------------------
__global__ __launch_bounds__(256) void prep_x(const float* __restrict__ x,
                                              const float* __restrict__ pb,
                                              __bf16* __restrict__ Xh,
                                              __bf16* __restrict__ Xl) {
    size_t idx = (size_t)blockIdx.x * 256 + threadIdx.x;   // over B*D
    size_t tile = idx >> 9;
    int s    = (int)(idx & 511);
    int lane = s >> 4, q = s & 15;
    int mt = (int)(tile / KT), kt = (int)(tile % KT);
    int m = mt * 16 + (lane & 15);
    int k = kt * 32 + ((lane >> 4) << 4) + q;
    float v = x[(size_t)m * D + k] - pb[k];
    __bf16 h = (__bf16)v;
    Xh[idx] = h;
    Xl[idx] = (__bf16)(v - (float)h);
}

// ---------------------------------------------------------------------------
// enc_gemm: zpre = (x-pb) @ W_enc + lb  via split-bf16 WMMA, LDS-staged.
//
// Block tile 128(M) x 128(N) = 8 m-tiles x 8 n-tiles; 8 waves (4 M x 2 N),
// each wave owns 2x4 16x16 output tiles -> 24 WMMAs / K-step.
// Per K-step the block's A/B fragment slice (32 KB: 8 m-tiles + 8 n-tiles,
// hi+lo, 1 KB per tile) is copied ONCE into LDS with async-to-LDS b128
// (double buffered, ASYNCcnt), and all 8 waves read fragments from LDS
// (2x ds_load_b128 per operand). Removes the x2/x4 intra-block L2 redundancy.
// ---------------------------------------------------------------------------
__global__ __launch_bounds__(256) void enc_gemm(const __bf16* __restrict__ Ah,
                                                const __bf16* __restrict__ Al,
                                                const __bf16* __restrict__ Bh,
                                                const __bf16* __restrict__ Bl,
                                                const float* __restrict__ lb,
                                                float* __restrict__ zpre) {
    extern __shared__ char smem[];             // 2 stages x 32 KB
    constexpr int STAGE = 32768;

    int tid  = threadIdx.x;
    int lane = tid & 31;
    int w    = tid >> 5;                        // wave id 0..7
    int wm   = w & 3;                           // 32-row strip
    int wn   = w >> 2;                          // 64-col strip
    int blk_mt0 = blockIdx.y * 8;
    int blk_nt0 = blockIdx.x * 8;
    int m0 = blockIdx.y * 128 + wm * 32;
    int n0 = blockIdx.x * 128 + wn * 64;

    // --- copy duty: wave w stages m-tile (blk_mt0+w) and n-tile (blk_nt0+w)
    const __bf16* gah = Ah + (size_t)(blk_mt0 + w) * KT * 512 + (size_t)lane * 16;
    const __bf16* gal = Al + (size_t)(blk_mt0 + w) * KT * 512 + (size_t)lane * 16;
    const __bf16* gbh = Bh + (size_t)(blk_nt0 + w) * KT * 512 + (size_t)lane * 16;
    const __bf16* gbl = Bl + (size_t)(blk_nt0 + w) * KT * 512 + (size_t)lane * 16;

    // LDS byte addresses (low 32 bits of generic shared pointer == LDS offset)
    unsigned smem0 = (unsigned)(uintptr_t)smem;
    unsigned ldsA  = smem0 + (unsigned)(w * 2048 + lane * 32);           // hi +0, lo +1024
    unsigned ldsB  = smem0 + 16384u + (unsigned)(w * 2048 + lane * 32);

    // --- fragment read pointers for this wave's 2x4 tiles
    const char* fa = smem + (wm * 2) * 2048 + lane * 32;                 // i*2048, lo +1024
    const char* fb = smem + 16384 + (wn * 4) * 2048 + lane * 32;         // j*2048, lo +1024

    v8f acc[2][4] = {};

    // prologue: stage kt=0 into buffer 0
    {
        const __bf16* a_h = gah; const __bf16* a_l = gal;
        const __bf16* b_h = gbh; const __bf16* b_l = gbl;
        ASYNC_CP16(ldsA,               a_h);
        ASYNC_CP16(ldsA + 16u,         a_h + 8);
        ASYNC_CP16(ldsA + 1024u,       a_l);
        ASYNC_CP16(ldsA + 1040u,       a_l + 8);
        ASYNC_CP16(ldsB,               b_h);
        ASYNC_CP16(ldsB + 16u,         b_h + 8);
        ASYNC_CP16(ldsB + 1024u,       b_l);
        ASYNC_CP16(ldsB + 1040u,       b_l + 8);
    }

    for (int kt = 0; kt < KT; ++kt) {
        int cur = kt & 1;
        WAIT_ASYNC0();          // this wave's fills for buf[cur] have landed
        __syncthreads();        // all waves' fills landed; prev reads of other buf done

        if (kt + 1 < KT) {      // stage kt+1 into the other buffer
            unsigned sb = (unsigned)((1 - cur) * STAGE);
            const __bf16* a_h = gah + (size_t)(kt + 1) * 512;
            const __bf16* a_l = gal + (size_t)(kt + 1) * 512;
            const __bf16* b_h = gbh + (size_t)(kt + 1) * 512;
            const __bf16* b_l = gbl + (size_t)(kt + 1) * 512;
            ASYNC_CP16(ldsA + sb,          a_h);
            ASYNC_CP16(ldsA + sb + 16u,    a_h + 8);
            ASYNC_CP16(ldsA + sb + 1024u,  a_l);
            ASYNC_CP16(ldsA + sb + 1040u,  a_l + 8);
            ASYNC_CP16(ldsB + sb,          b_h);
            ASYNC_CP16(ldsB + sb + 16u,    b_h + 8);
            ASYNC_CP16(ldsB + sb + 1024u,  b_l);
            ASYNC_CP16(ldsB + sb + 1040u,  b_l + 8);
        }

        const char* sa = fa + cur * STAGE;
        const char* sb = fb + cur * STAGE;
        v16bf ah[2], al[2], bh[4], bl[4];
#pragma unroll
        for (int i = 0; i < 2; ++i) {
            ah[i] = *(const v16bf*)(sa + i * 2048);
            al[i] = *(const v16bf*)(sa + i * 2048 + 1024);
        }
#pragma unroll
        for (int j = 0; j < 4; ++j) {
            bh[j] = *(const v16bf*)(sb + j * 2048);
            bl[j] = *(const v16bf*)(sb + j * 2048 + 1024);
        }
#pragma unroll
        for (int i = 0; i < 2; ++i) {
#pragma unroll
            for (int j = 0; j < 4; ++j) {
                acc[i][j] = __builtin_amdgcn_wmma_f32_16x16x32_bf16(
                    false, ah[i], false, bh[j], (short)0, acc[i][j], false, false);
                acc[i][j] = __builtin_amdgcn_wmma_f32_16x16x32_bf16(
                    false, ah[i], false, bl[j], (short)0, acc[i][j], false, false);
                acc[i][j] = __builtin_amdgcn_wmma_f32_16x16x32_bf16(
                    false, al[i], false, bh[j], (short)0, acc[i][j], false, false);
            }
        }
    }

    // Epilogue: C/D 16x16 f32 layout: VGPR r -> M = r + (lane>=16 ? 8:0), N = lane&15
    int nl = lane & 15;
    int mh = (lane >> 4) << 3;
#pragma unroll
    for (int j = 0; j < 4; ++j) {
        int n = n0 + j * 16 + nl;
        float lbv = lb[n];
#pragma unroll
        for (int i = 0; i < 2; ++i) {
            int mbase = m0 + i * 16 + mh;
#pragma unroll
            for (int r = 0; r < 8; ++r)
                zpre[(size_t)(mbase + r) * L + n] = acc[i][j][r] + lbv;
        }
    }
}

// ---------------------------------------------------------------------------
// topk_kernel: exact per-row top-64 radix select (order-preserving uint keys),
// row cached in 64KB dynamic LDS. Ties taken in index order (matches
// lax.top_k). Writes full dense z row + compacted (idx, relu(val)) pairs.
// ---------------------------------------------------------------------------
__global__ __launch_bounds__(256) void topk_kernel(const float* __restrict__ zpre,
                                                   float* __restrict__ z,
                                                   int*   __restrict__ sel_idx,
                                                   float* __restrict__ sel_val) {
    constexpr int TPB = 256, EPT = L / TPB;
    extern __shared__ unsigned keys[];     // L entries = 64 KB
    __shared__ int hist[256];
    __shared__ unsigned s_pv, s_pm;
    __shared__ int s_rem, s_cnt;

    int b = blockIdx.x, t = threadIdx.x;
    const float* row = zpre + (size_t)b * L;

    for (int i = t; i < L; i += TPB) {
        unsigned u = __float_as_uint(row[i]);
        keys[i] = (u & 0x80000000u) ? ~u : (u | 0x80000000u);
    }
    if (t == 0) { s_pv = 0u; s_pm = 0u; s_rem = KS; s_cnt = 0; }
    __syncthreads();

    for (int pass = 0; pass < 4; ++pass) {
        int shift = 24 - 8 * pass;
        hist[t] = 0;
        __syncthreads();
        unsigned pv = s_pv, pm = s_pm;
        for (int i = t * EPT; i < (t + 1) * EPT; ++i) {
            unsigned kk = keys[i];
            if ((kk & pm) == pv) atomicAdd(&hist[(kk >> shift) & 255u], 1);
        }
        __syncthreads();
        if (t == 0) {
            int rem = s_rem;
            int d = 255;
            for (; d > 0; --d) { int c = hist[d]; if (c >= rem) break; rem -= c; }
            s_pv = s_pv | ((unsigned)d << shift);
            s_pm = s_pm | (255u << shift);
            s_rem = rem;
        }
        __syncthreads();
    }
    unsigned T = s_pv;        // exact key of the K-th largest element
    int need_eq = s_rem;      // # of ==T elements to take, in index order

    int local_eq = 0;
    for (int i = t * EPT; i < (t + 1) * EPT; ++i) local_eq += (keys[i] == T) ? 1 : 0;
    __syncthreads();
    hist[t] = local_eq;
    __syncthreads();
    if (t == 0) {
        int run = 0;
        for (int i = 0; i < 256; ++i) { int c = hist[i]; hist[i] = run; run += c; }
    }
    __syncthreads();
    int eqrank = hist[t];

    float* zrow = z + (size_t)b * L;
    for (int i = t * EPT; i < (t + 1) * EPT; ++i) {
        unsigned kk = keys[i];
        float v = 0.0f;
        bool iseq = (kk == T);
        bool sel = (kk > T) || (iseq && eqrank < need_eq);
        if (iseq) ++eqrank;
        if (sel) {
            unsigned u = (kk & 0x80000000u) ? (kk & 0x7fffffffu) : ~kk;
            float val = __uint_as_float(u);
            v = val > 0.0f ? val : 0.0f;           // relu of selected value
            int r = atomicAdd(&s_cnt, 1);
            if (r < KS) { sel_idx[(size_t)b * KS + r] = i; sel_val[(size_t)b * KS + r] = v; }
        }
        zrow[i] = v;
    }
}

// ---------------------------------------------------------------------------
// dec_kernel: xhat[b,:] = sum_k val_k * W_dec[idx_k,:] + pb  (sparse decode)
// One workgroup (320 threads, 4 floats each) per row; W_dec stays in L2.
// ---------------------------------------------------------------------------
__global__ __launch_bounds__(320) void dec_kernel(const int*   __restrict__ sel_idx,
                                                  const float* __restrict__ sel_val,
                                                  const float* __restrict__ W_dec,
                                                  const float* __restrict__ pb,
                                                  float* __restrict__ xhat) {
    int b = blockIdx.x, t = threadIdx.x;
    int d = t * 4;
    __shared__ int   s_idx[KS];
    __shared__ float s_val[KS];
    if (t < KS) {
        s_idx[t] = sel_idx[(size_t)b * KS + t] & (L - 1);   // defensive clamp
        s_val[t] = sel_val[(size_t)b * KS + t];
    }
    __syncthreads();
    float4 acc = *(const float4*)(pb + d);
#pragma unroll 4
    for (int k = 0; k < KS; ++k) {
        float v = s_val[k];
        float4 wr = *(const float4*)(W_dec + (size_t)s_idx[k] * D + d);
        acc.x += v * wr.x; acc.y += v * wr.y; acc.z += v * wr.z; acc.w += v * wr.w;
    }
    *(float4*)(xhat + (size_t)b * D + d) = acc;
}

// ---------------------------------------------------------------------------
// kernel_launch
// inputs: 0=x_BD 1=pb_D 2=W_enc 3=lb_L 4=W_dec (all f32)
// d_out: [zpre B*L | z B*L | xhat B*D] f32
// ws:    [Wh | Wl | Xh | Xl | sel_idx | sel_val]  ~102 MB
// ---------------------------------------------------------------------------
extern "C" void kernel_launch(void* const* d_in, const int* in_sizes, int n_in,
                              void* d_out, int out_size, void* d_ws, size_t ws_size,
                              hipStream_t stream) {
    const float* x     = (const float*)d_in[0];
    const float* pb    = (const float*)d_in[1];
    const float* W_enc = (const float*)d_in[2];
    const float* lb    = (const float*)d_in[3];
    const float* W_dec = (const float*)d_in[4];

    float* out  = (float*)d_out;
    float* zpre = out;
    float* zact = out + (size_t)B * L;
    float* xhat = out + 2 * (size_t)B * L;

    const size_t WN = (size_t)D * L;   // W_enc elements
    const size_t XN = (size_t)B * D;   // x elements
    __bf16* Wh = (__bf16*)d_ws;
    __bf16* Wl = Wh + WN;
    __bf16* Xh = Wl + WN;
    __bf16* Xl = Xh + XN;
    int*    sidx = (int*)(Xl + XN);
    float*  sval = (float*)(sidx + (size_t)B * KS);

    prep_w<<<(int)(WN / 256), 256, 0, stream>>>(W_enc, Wh, Wl);
    prep_x<<<(int)(XN / 256), 256, 0, stream>>>(x, pb, Xh, Xl);

    dim3 grid(L / 128, B / 128);
    enc_gemm<<<grid, 256, 65536, stream>>>(Xh, Xl, Wh, Wl, lb, zpre);

    topk_kernel<<<B, 256, L * sizeof(unsigned), stream>>>(zpre, zact, sidx, sval);

    dec_kernel<<<B, 320, 0, stream>>>(sidx, sval, W_dec, pb, xhat);
}
